// NonKeyC1_81363860455651
// MI455X (gfx1250) — matrix operand order repaired and checked by reading, not compiled, not run on hardware
//
#include <hip/hip_runtime.h>

// ---------------------------------------------------------------------------
// MI455X (gfx1250) implementation.
//  * All convs -> im2col GEMM on v_wmma_f32_16x16x32_bf16 (fp32 accum).
//    Wave tile 32Mx32N (4 WMMA / K-step), block tile 64 pixels x 128 chans.
//  * Weight tiles staged LDS via global_load_async_to_lds_b128 (ASYNCcnt).
//  * BN folded into bf16 weights at prep time; activations kept in bf16.
//  * Fused bilinear(align_corners)+log_softmax for the 713x713 output.
// ---------------------------------------------------------------------------

typedef __attribute__((ext_vector_type(16))) __bf16 v16bf;
typedef __attribute__((ext_vector_type(8)))  float  v8f;

#define HW   8100
#define IMG  90
#define OUTHW 713
#define NCLS 150

__device__ __forceinline__ __bf16 f2bf(float f) {
    union { float f; unsigned u; } a; a.f = f;
    unsigned u = a.u + 0x7FFFu + ((a.u >> 16) & 1u);   // round-to-nearest-even
    union { unsigned short s; __bf16 b; } c; c.s = (unsigned short)(u >> 16);
    return c.b;
}
__device__ __forceinline__ float bf2f(__bf16 b) {
    union { unsigned short s; __bf16 b; } c; c.b = b;
    union { unsigned u; float f; } a; a.u = ((unsigned)c.s) << 16;
    return a.f;
}

// ---------------------------------------------------------------------------
// Weight prep: fold BN (g,b,m,v) into per-channel scale/bias, repack OIHW ->
// [CoutPad][tap][Cin] bf16 so GEMM-K (tap*Cin+ci) is contiguous per channel.
// ---------------------------------------------------------------------------
__global__ void prep_weights(const float* __restrict__ w, const float* __restrict__ g,
                             const float* __restrict__ bb, const float* __restrict__ mm,
                             const float* __restrict__ vv,
                             __bf16* __restrict__ Wb, float* __restrict__ biasOut,
                             int Cin, int taps, int CoutReal, int CoutPad)
{
    const int Ktot = Cin * taps;
    const long long total = (long long)CoutPad * Ktot;
    for (long long idx = (long long)blockIdx.x * blockDim.x + threadIdx.x;
         idx < total; idx += (long long)gridDim.x * blockDim.x) {
        int o  = (int)(idx / Ktot);
        int k  = (int)(idx - (long long)o * Ktot);
        int tap = k / Cin;
        int ci  = k - tap * Cin;
        float scale = 1.f, bias = 0.f;
        if (o < CoutReal) {
            if (g) { scale = g[o] * rsqrtf(vv[o] + 1e-5f); bias = bb[o] - mm[o] * scale; }
            else   { bias = bb[o]; }
            float wv = w[((size_t)o * Cin + ci) * taps + tap];
            Wb[idx] = f2bf(wv * scale);
        } else {
            Wb[idx] = f2bf(0.f);
        }
        if (k == 0) biasOut[o] = bias;
    }
}

__global__ void cast_f32_bf16(const float* __restrict__ src, __bf16* __restrict__ dst, int n)
{
    int i = blockIdx.x * blockDim.x + threadIdx.x;
    if (i < n) dst[i] = f2bf(src[i]);
}

// ---------------------------------------------------------------------------
// Generic conv (3x3 pad 1 / 1x1) as WMMA GEMM.
//  Block = 256 threads (8 waves). Tile: M=64 pixels x N=128 channels.
//  Wave (mg=wave&1, ng=wave>>1): 32x32 output tile = 4 wmma accumulators,
//  reusing 2 A-fragments x 2 B-fragments per K=32 step (K never straddles a
//  tap since Cin % 32 == 0 for every layer).
//  B tile (contiguous bf16 weights) is staged into LDS with
//  global_load_async_to_lds_b128 and fenced with s_wait_asynccnt.
//  LDS rows have 80B stride so per-lane ds_load_b128 fragment reads are
//  bank-conflict-free across all 64 banks.
// ---------------------------------------------------------------------------
__global__ __launch_bounds__(256)
void conv_wmma(const __bf16* __restrict__ X,      // [Cin][8100]
               const __bf16* __restrict__ W,      // [256][taps*Cin]
               const float*  __restrict__ bias,   // [256]
               float*  __restrict__ outF,         // optional f32 NCHW out
               __bf16* __restrict__ outB,         // optional bf16 NCHW out
               int Cin, int CoutReal, int is3x3, int relu)
{
    const int Ktot  = Cin * (is3x3 ? 9 : 1);
    const int mBase = blockIdx.x * 64;
    const int nBase = blockIdx.y * 128;
    const int tid   = threadIdx.x;
    const int lane  = tid & 31;
    const int wave  = tid >> 5;
    const int mg    = wave & 1;        // 0..1 : 32-pixel M tile
    const int ng    = wave >> 1;       // 0..3 : 32-channel N tile

    __shared__ __bf16 As[64 * 40];     // 64 pixels x 32 K, stride 40 halves
    __shared__ __bf16 Bs[128 * 40];    // 128 chans x 32 K

    v8f a00 = {0.f,0.f,0.f,0.f,0.f,0.f,0.f,0.f};
    v8f a01 = a00, a10 = a00, a11 = a00;

    // A staging map: 4 threads per pixel row, 8 consecutive K each
    const int sRow = tid >> 2;             // 0..63
    const int sK   = (tid & 3) * 8;        // 0,8,16,24
    const int pA   = mBase + sRow;
    const int ph   = pA / IMG;
    const int pw   = pA - ph * IMG;

    // B staging map: 2 threads per channel row, 16 consecutive K (32B) each
    const int sRowB = tid >> 1;            // 0..127
    const int sKB   = (tid & 1) * 16;      // 0,16
    const unsigned ldsB =
        (unsigned)(unsigned long long)&Bs[sRowB * 40 + sKB];

    const int hlf = lane >> 4;             // 0|1: half-wave
    const int l15 = lane & 15;

    for (int k0 = 0; k0 < Ktot; k0 += 32) {
        int tap = 0, ci0 = k0;
        if (is3x3) { tap = k0 / Cin; ci0 = k0 - tap * Cin; }
        const int dh = is3x3 ? (tap / 3 - 1) : 0;
        const int dw = is3x3 ? (tap % 3 - 1) : 0;

        // ---- stage B asynchronously: 2x b128 straight to LDS ----
        {
            const __bf16* srcB = W + (size_t)(nBase + sRowB) * Ktot + k0 + sKB;
            unsigned long long ga = (unsigned long long)srcB;
            asm volatile("global_load_async_to_lds_b128 %0, %1, off"
                         :: "v"(ldsB), "v"(ga) : "memory");
            asm volatile("global_load_async_to_lds_b128 %0, %1, off offset:16"
                         :: "v"(ldsB), "v"(ga) : "memory");
        }

        // ---- stage A (im2col gather, zero-fill at borders) ----
        {
            const int hh = ph + dh, ww = pw + dw;
            const bool ok = (pA < HW) && ((unsigned)hh < (unsigned)IMG)
                                      && ((unsigned)ww < (unsigned)IMG);
            const __bf16* src = X + (size_t)(ci0 + sK) * HW + (hh * IMG + ww);
            __bf16* dst = &As[sRow * 40 + sK];
            if (ok) __builtin_prefetch(src + (size_t)32 * HW, 0, 3); // next K chunk
            #pragma unroll
            for (int kk = 0; kk < 8; ++kk)
                dst[kk] = ok ? src[(size_t)kk * HW] : f2bf(0.f);
        }

        asm volatile("s_wait_asynccnt 0x0" ::: "memory");
        __syncthreads();

        // ---- fragments per ISA 16-bit A(16x32)/B(32x16) layouts ----
        union { v16bf v; uint4 q[2]; } af0, af1, bf0, bf1;
        {
            const __bf16* ar0 = &As[(mg * 32 + l15) * 40];
            af0.q[0] = *(const uint4*)&ar0[hlf ? 8  : 0];
            af0.q[1] = *(const uint4*)&ar0[hlf ? 24 : 16];
            const __bf16* ar1 = &As[(mg * 32 + 16 + l15) * 40];
            af1.q[0] = *(const uint4*)&ar1[hlf ? 8  : 0];
            af1.q[1] = *(const uint4*)&ar1[hlf ? 24 : 16];
        }
        {
            const __bf16* b0 = &Bs[(ng * 32 + l15) * 40 + (hlf ? 8 : 0)];
            bf0.q[0] = *(const uint4*)&b0[0];
            bf0.q[1] = *(const uint4*)&b0[16];
            const __bf16* b1 = &Bs[(ng * 32 + 16 + l15) * 40 + (hlf ? 8 : 0)];
            bf1.q[0] = *(const uint4*)&b1[0];
            bf1.q[1] = *(const uint4*)&b1[16];
        }
        a00 = __builtin_amdgcn_wmma_f32_16x16x32_bf16(false, af0.v, false, bf0.v,
                                                      (short)0, a00, false, false);
        a01 = __builtin_amdgcn_wmma_f32_16x16x32_bf16(false, af0.v, false, bf1.v,
                                                      (short)0, a01, false, false);
        a10 = __builtin_amdgcn_wmma_f32_16x16x32_bf16(false, af1.v, false, bf0.v,
                                                      (short)0, a10, false, false);
        a11 = __builtin_amdgcn_wmma_f32_16x16x32_bf16(false, af1.v, false, bf1.v,
                                                      (short)0, a11, false, false);
        __syncthreads();
    }

    // ---- epilogue: C/D layout = VGPR r -> row M=r+8*hlf, col N=l15 ----
    const int oc0 = nBase + ng * 32 + l15;
    const int oc1 = oc0 + 16;
    const float b0 = bias[oc0], b1 = bias[oc1];
    #pragma unroll
    for (int r = 0; r < 8; ++r) {
        const int p0 = mBase + mg * 32 + r + (hlf ? 8 : 0);
        const int p1 = p0 + 16;
        float v00 = a00[r] + b0, v01 = a01[r] + b1;
        float v10 = a10[r] + b0, v11 = a11[r] + b1;
        if (relu) {
            v00 = fmaxf(v00, 0.f); v01 = fmaxf(v01, 0.f);
            v10 = fmaxf(v10, 0.f); v11 = fmaxf(v11, 0.f);
        }
        if (p0 < HW) {
            if (oc0 < CoutReal) {
                if (outF) outF[(size_t)oc0 * HW + p0] = v00;
                if (outB) outB[(size_t)oc0 * HW + p0] = f2bf(v00);
            }
            if (oc1 < CoutReal) {
                if (outF) outF[(size_t)oc1 * HW + p0] = v01;
                if (outB) outB[(size_t)oc1 * HW + p0] = f2bf(v01);
            }
        }
        if (p1 < HW) {
            if (oc0 < CoutReal) {
                if (outF) outF[(size_t)oc0 * HW + p1] = v10;
                if (outB) outB[(size_t)oc0 * HW + p1] = f2bf(v10);
            }
            if (oc1 < CoutReal) {
                if (outF) outF[(size_t)oc1 * HW + p1] = v11;
                if (outB) outB[(size_t)oc1 * HW + p1] = f2bf(v11);
            }
        }
    }
}

// ---------------------------------------------------------------------------
// 9x9 local attention weighting: out[c,p] = sum_t hk_pad[c, p+off(t)] * att[p,t]
// Block = 8 channels (waves) x 32 pixels (lanes) -> coalesced bf16 reads.
// ---------------------------------------------------------------------------
__global__ __launch_bounds__(256)
void f_weighting(const __bf16* __restrict__ hk, const float* __restrict__ atten,
                 __bf16* __restrict__ out)
{
    const int lane = threadIdx.x & 31, wv = threadIdx.x >> 5;
    const int p = blockIdx.x * 32 + lane;
    const int c = blockIdx.y * 8 + wv;
    if (p >= HW) return;
    const int h = p / IMG, w = p - h * IMG;
    const float* att = atten + (size_t)p * 81;
    const __bf16* xc = hk + (size_t)c * HW;
    float acc = 0.f;
    #pragma unroll
    for (int i = 0; i < 9; ++i) {
        const int hh = h + i - 4;
        #pragma unroll
        for (int j = 0; j < 9; ++j) {
            const int ww = w + j - 4;
            float xv = (((unsigned)hh < (unsigned)IMG) && ((unsigned)ww < (unsigned)IMG))
                       ? bf2f(xc[hh * IMG + ww]) : 0.f;
            acc = fmaf(xv, att[i * 9 + j], acc);
        }
    }
    out[(size_t)c * HW + p] = f2bf(acc);
}

// ---------------------------------------------------------------------------
// Fused bilinear (align_corners=True) 90->713 + channel log_softmax.
// Logit tensor (150x8100, 4.9MB) lives in L2; recompute interp across the
// 3 passes (max, sumexp, write) to keep register pressure bounded.
// ---------------------------------------------------------------------------
__global__ __launch_bounds__(256)
void resize_logsoftmax(const float* __restrict__ logits, float* __restrict__ out)
{
    const int idx = blockIdx.x * blockDim.x + threadIdx.x;
    if (idx >= OUTHW * OUTHW) return;
    const int oy = idx / OUTHW, ox = idx - oy * OUTHW;
    const float s = 89.0f / 712.0f;
    const float fy = oy * s, fx = ox * s;
    int y0 = (int)fy; if (y0 > 88) y0 = 88;
    int x0 = (int)fx; if (x0 > 88) x0 = 88;
    const float wy = fy - (float)y0, wx = fx - (float)x0;
    const int y1 = (y0 + 1 < 89) ? y0 + 1 : 89;
    const int x1 = (x0 + 1 < 89) ? x0 + 1 : 89;
    const int i00 = y0 * IMG + x0, i01 = y0 * IMG + x1;
    const int i10 = y1 * IMG + x0, i11 = y1 * IMG + x1;
    const float w00 = (1.f - wy) * (1.f - wx), w01 = (1.f - wy) * wx;
    const float w10 = wy * (1.f - wx),         w11 = wy * wx;

    float mx = -1e30f;
    for (int c = 0; c < NCLS; ++c) {
        const float* L = logits + (size_t)c * HW;
        float v = L[i00]*w00 + L[i01]*w01 + L[i10]*w10 + L[i11]*w11;
        mx = fmaxf(mx, v);
    }
    float sum = 0.f;
    for (int c = 0; c < NCLS; ++c) {
        const float* L = logits + (size_t)c * HW;
        float v = L[i00]*w00 + L[i01]*w01 + L[i10]*w10 + L[i11]*w11;
        sum += __expf(v - mx);
    }
    const float lse = mx + __logf(sum);
    for (int c = 0; c < NCLS; ++c) {
        const float* L = logits + (size_t)c * HW;
        float v = L[i00]*w00 + L[i01]*w01 + L[i10]*w10 + L[i11]*w11;
        out[(size_t)c * (OUTHW * OUTHW) + idx] = v - lse;
    }
}

// ---------------------------------------------------------------------------
extern "C" void kernel_launch(void* const* d_in, const int* in_sizes, int n_in,
                              void* d_out, int out_size, void* d_ws, size_t ws_size,
                              hipStream_t stream)
{
    (void)in_sizes; (void)n_in; (void)out_size; (void)ws_size;
    // setup_inputs() dict order (nested params flattened in insertion order):
    const float* low   = (const float*)d_in[0];   // (1,1024,90,90)
    const float* high  = (const float*)d_in[1];   // (1,2048,90,90)
    const float* atten = (const float*)d_in[2];   // (1,90,90,81)
    // params: cbr, lowcbr, lowcbr2, lowcbr3, cbr3, cbr4 each {w,g,b,m,v}
    const float* W_[6];  const float* G_[6]; const float* B_[6];
    const float* M_[6];  const float* V_[6];
    for (int l = 0; l < 6; ++l) {
        W_[l] = (const float*)d_in[3 + 5*l + 0];
        G_[l] = (const float*)d_in[3 + 5*l + 1];
        B_[l] = (const float*)d_in[3 + 5*l + 2];
        M_[l] = (const float*)d_in[3 + 5*l + 3];
        V_[l] = (const float*)d_in[3 + 5*l + 4];
    }
    const float* last_w = (const float*)d_in[33];
    const float* last_b = (const float*)d_in[34];

    // workspace bump allocator
    char* p = (char*)d_ws;
    auto alloc = [&](size_t bytes) { char* r = p; p += (bytes + 255) & ~(size_t)255; return r; };

    const int Cin_[6]  = {2048, 1024, 256, 256, 512, 256};
    __bf16* Wb_[6]; float* bias_[6];
    for (int l = 0; l < 6; ++l) {
        Wb_[l]   = (__bf16*)alloc((size_t)256 * Cin_[l] * 9 * 2);
        bias_[l] = (float*)alloc(1024);
    }
    __bf16* WbLast   = (__bf16*)alloc((size_t)256 * 256 * 2);  // padded to 256 rows
    float*  biasLast = (float*)alloc(1024);

    __bf16* bufA   = (__bf16*)alloc((size_t)2048 * HW * 2);  // cast inputs (reused)
    __bf16* buf_a  = (__bf16*)alloc((size_t)256 * HW * 2);   // hk, later cbr4 out
    __bf16* buf_b  = (__bf16*)alloc((size_t)256 * HW * 2);   // y1, later cbr3-bf16
    __bf16* buf_c  = (__bf16*)alloc((size_t)256 * HW * 2);   // y2
    __bf16* concat = (__bf16*)alloc((size_t)512 * HW * 2);   // [fw | y3]
    float*  logits = (float*)alloc((size_t)192 * HW * 4);

    float* out1F   = (float*)d_out;                          // 256*8100
    float* outBig  = out1F + (size_t)256 * HW;               // 150*713*713

    // --- weight prep (BN fold + bf16 repack) ---
    for (int l = 0; l < 6; ++l)
        prep_weights<<<1024, 256, 0, stream>>>(W_[l], G_[l], B_[l], M_[l], V_[l],
                                               Wb_[l], bias_[l], Cin_[l], 9, 256, 256);
    prep_weights<<<256, 256, 0, stream>>>(last_w, nullptr, last_b, nullptr, nullptr,
                                          WbLast, biasLast, 256, 1, 150, 256);

    const int gM = (HW + 63) / 64;   // 127
    dim3 blk(256);
    dim3 grd(gM, 2);                 // 128-channel N blocks, Cout padded to 256

    // --- high-key path: cbr -> f_weighting ---
    cast_f32_bf16<<<(2048 * HW + 255) / 256, 256, 0, stream>>>(high, bufA, 2048 * HW);
    conv_wmma<<<grd, blk, 0, stream>>>(bufA, Wb_[0], bias_[0],
                                       nullptr, buf_a, 2048, 256, 1, 1);
    f_weighting<<<dim3((HW + 31) / 32, 32), blk, 0, stream>>>(buf_a, atten, concat);

    // --- low path: lowcbr -> lowcbr2 -> lowcbr3 ---
    cast_f32_bf16<<<(1024 * HW + 255) / 256, 256, 0, stream>>>(low, bufA, 1024 * HW);
    conv_wmma<<<grd, blk, 0, stream>>>(bufA,  Wb_[1], bias_[1],
                                       nullptr, buf_b, 1024, 256, 1, 1);
    conv_wmma<<<grd, blk, 0, stream>>>(buf_b, Wb_[2], bias_[2],
                                       nullptr, buf_c, 256, 256, 1, 1);
    conv_wmma<<<grd, blk, 0, stream>>>(buf_c, Wb_[3], bias_[3],
                                       nullptr, concat + (size_t)256 * HW, 256, 256, 1, 1);

    // --- head: cbr3 (out1) -> cbr4 -> 1x1 -> resize+log_softmax ---
    conv_wmma<<<grd, blk, 0, stream>>>(concat, Wb_[4], bias_[4],
                                       out1F, buf_b, 512, 256, 1, 1);
    conv_wmma<<<grd, blk, 0, stream>>>(buf_b, Wb_[5], bias_[5],
                                       nullptr, buf_a, 256, 256, 1, 1);
    conv_wmma<<<grd, blk, 0, stream>>>(buf_a, WbLast, biasLast,
                                       logits, nullptr, 256, 150, 0, 0);
    resize_logsoftmax<<<(OUTHW * OUTHW + 255) / 256, 256, 0, stream>>>(logits, outBig);
}
